// WriteHead_66623532695698
// MI455X (gfx1250) — matrix operand order repaired and checked by reading, not compiled, not run on hardware
//
#include <hip/hip_runtime.h>
#include <hip/hip_bf16.h>
#include <math.h>

#define BB 64
#define NN 1024
#define WW 128
#define EPSF 1e-8f

typedef __attribute__((ext_vector_type(16))) _Float16 v16h;
typedef __attribute__((ext_vector_type(8)))  float    v8f;
typedef __attribute__((ext_vector_type(4)))  float    v4f;

// ---------------- block reduction helpers (256 threads) ----------------
__device__ __forceinline__ float block_reduce_sum(float v, float* red, int tid) {
    red[tid] = v;
    __syncthreads();
    for (int s = 128; s > 0; s >>= 1) {
        if (tid < s) red[tid] += red[tid + s];
        __syncthreads();
    }
    float r = red[0];
    __syncthreads();
    return r;
}

__device__ __forceinline__ float block_reduce_max(float v, float* red, int tid) {
    red[tid] = v;
    __syncthreads();
    for (int s = 128; s > 0; s >>= 1) {
        if (tid < s) red[tid] = fmaxf(red[tid], red[tid + s]);
        __syncthreads();
    }
    float r = red[0];
    __syncthreads();
    return r;
}

// ---------------- kernel 1: content weights + write weights + precedence ----
// One block per batch. 8 wave32s. WMMA computes sim[n] = dot(memory[b,n,:], key[b,:])
// with norms folded out afterwards (identical to normalize-then-dot).
__global__ __launch_bounds__(256) void head_kernel(
    const float* __restrict__ w_key,    // (B,W)
    const float* __restrict__ w_beta,   // (B,1)
    const float* __restrict__ a_gate,   // (B,1)
    const float* __restrict__ w_gate,   // (B,1)
    const float* __restrict__ alloc,    // (B,N)
    const float* __restrict__ memory,   // (B,N,W)
    const float* __restrict__ prec,     // (B,N)
    float* __restrict__ out_w,          // (B,N)
    float* __restrict__ out_prec)       // (B,N)
{
    __shared__ _Float16 s_keyh[WW];
    __shared__ float    s_sim[NN];
    __shared__ float    s_ssq[NN];
    __shared__ float    s_red[256];

    const int b    = blockIdx.x;
    const int tid  = threadIdx.x;
    const int lane = tid & 31;
    const int wave = tid >> 5;

    // ---- phase A: key -> f16 in LDS, key norm, scalar gates ----
    float ksq = 0.0f;
    if (tid < WW) {
        float kv = w_key[b * WW + tid];
        s_keyh[tid] = (_Float16)kv;
        ksq = kv * kv;
    }
    const float normk = sqrtf(block_reduce_sum(ksq, s_red, tid));

    const float braw = w_beta[b];
    const float beta = 1.0f + ((braw > 20.0f) ? braw : log1pf(expf(braw)));
    const float ag   = 1.0f / (1.0f + expf(-a_gate[b]));
    const float wg   = 1.0f / (1.0f + expf(-w_gate[b]));

    // ---- phase B: WMMA similarity + row sum-of-squares ----
    // A layout (16-bit 16x32): lanes 0-15 = M rows, elems 0..7 -> K = half*8+0..7,
    // elems 8..15 -> K = 16+half*8+0..7 (half = lane>=16).
    // B layout (32x16): same K order as A: lanes 0-15 hold K=0..7,16..23, lanes 16-31 K=8..15,24..31;
    // key broadcast to all 16 columns -> every D column equals the sim vector.
    const int half = lane >> 4;
    const int mrow = lane & 15;

    for (int t = wave; t < NN / 16; t += 8) {
        const int row = t * 16 + mrow;
        const float* mrp = memory + ((size_t)b * NN + row) * WW;

        v8f  acc = {};
        float ssq = 0.0f;

        for (int kk = 0; kk < WW; kk += 32) {
            v16h a, bf;
            #pragma unroll
            for (int j = 0; j < 8; ++j) {
                float f0 = mrp[kk + half * 8 + j];
                float f1 = mrp[kk + 16 + half * 8 + j];
                a[j]     = (_Float16)f0;
                a[8 + j] = (_Float16)f1;
                ssq += f0 * f0 + f1 * f1;
            }
            #pragma unroll
            for (int e = 0; e < 16; ++e)
                bf[e] = s_keyh[kk + ((e < 8) ? (half * 8 + e) : (16 + half * 8 + (e - 8)))];

            acc = __builtin_amdgcn_wmma_f32_16x16x32_f16(
                false, a, false, bf, (short)0, acc, false, false);
        }

        // full row sum-of-squares: lane m covers K{0-7,16-23,...}, lane m+16 the rest
        float ssq_full = ssq + __shfl_xor(ssq, 16, 32);
        if (lane < 16) s_ssq[t * 16 + mrow] = ssq_full;

        // D: VGPR r holds row M=r (lanes 0-15) / M=r+8 (lanes 16-31); columns identical
        if (lane == 0) {
            #pragma unroll
            for (int r = 0; r < 8; ++r) s_sim[t * 16 + r] = acc[r];
        } else if (lane == 16) {
            #pragma unroll
            for (int r = 0; r < 8; ++r) s_sim[t * 16 + 8 + r] = acc[r];
        }
    }
    __syncthreads();

    // ---- phase C: scale + softmax + gates + precedence ----
    float vals[4];
    float lmax = -INFINITY;
    #pragma unroll
    for (int c = 0; c < 4; ++c) {
        int n = tid * 4 + c;
        float scale = beta / ((sqrtf(s_ssq[n]) + EPSF) * (normk + EPSF));
        float s = s_sim[n] * scale;
        vals[c] = s;
        lmax = fmaxf(lmax, s);
    }
    const float gmax = block_reduce_max(lmax, s_red, tid);

    float lsum = 0.0f;
    #pragma unroll
    for (int c = 0; c < 4; ++c) {
        vals[c] = expf(vals[c] - gmax);
        lsum += vals[c];
    }
    const float gsum = block_reduce_sum(lsum, s_red, tid);
    const float inv_gsum = 1.0f / gsum;

    float wv[4];
    float lw = 0.0f;
    #pragma unroll
    for (int c = 0; c < 4; ++c) {
        int n = tid * 4 + c;
        float cw = vals[c] * inv_gsum;
        float ww = wg * (ag * alloc[b * NN + n] + (1.0f - ag) * cw);
        wv[c] = ww;
        out_w[b * NN + n] = ww;
        lw += ww;
    }
    const float wsum = block_reduce_sum(lw, s_red, tid);

    #pragma unroll
    for (int c = 0; c < 4; ++c) {
        int n = tid * 4 + c;
        out_prec[b * NN + n] = (1.0f - wsum) * prec[b * NN + n] + wv[c];
    }
}

// ---------------- kernel 2: temporal link update (bandwidth-dominant) -------
// link[b,i,j] = (1 - w_i - w_j)*L[b,i,j] + w_i*p_j, diagonal zeroed.
// One block per (b,i) row; 256 threads x float4 = 1024 columns.
// Nontemporal on the 268MB+268MB stream so the hot w/p rows stay cached.
__global__ __launch_bounds__(256) void link_kernel(
    const float* __restrict__ link_in,  // (B,N,N)
    const float* __restrict__ ww,       // (B,N) -- written by head_kernel
    const float* __restrict__ prec,     // (B,N)
    float* __restrict__ link_out)       // (B,N,N)
{
    const int b = blockIdx.x >> 10;
    const int i = blockIdx.x & (NN - 1);
    const float wi = ww[b * NN + i];

    const int j4 = threadIdx.x * 4;
    const size_t base = ((size_t)b * NN + i) * NN + j4;

    v4f L  = __builtin_nontemporal_load((const v4f*)(link_in + base));
    v4f wj = *(const v4f*)(ww   + b * NN + j4);
    v4f pj = *(const v4f*)(prec + b * NN + j4);

    v4f r;
    #pragma unroll
    for (int c = 0; c < 4; ++c)
        r[c] = (1.0f - wi - wj[c]) * L[c] + wi * pj[c];

    if (i >= j4 && i < j4 + 4) r[i - j4] = 0.0f;

    __builtin_nontemporal_store(r, (v4f*)(link_out + base));
}

// ---------------- host entry ----------------
extern "C" void kernel_launch(void* const* d_in, const int* in_sizes, int n_in,
                              void* d_out, int out_size, void* d_ws, size_t ws_size,
                              hipStream_t stream) {
    const float* w_key  = (const float*)d_in[0];
    const float* w_beta = (const float*)d_in[1];
    // d_in[2] e_vector: sigmoid result discarded by reference
    // d_in[3] w_vector: unused by reference
    const float* a_gate = (const float*)d_in[4];
    const float* w_gate = (const float*)d_in[5];
    const float* alloc  = (const float*)d_in[6];
    const float* memory = (const float*)d_in[7];
    const float* linkm  = (const float*)d_in[8];
    const float* prec   = (const float*)d_in[9];

    float* out      = (float*)d_out;
    float* out_w    = out;                                   // (B,N)
    float* out_mem  = out_w + (size_t)BB * NN;               // (B,N,W)
    float* out_link = out_mem + (size_t)BB * NN * WW;        // (B,N,N)
    float* out_prec = out_link + (size_t)BB * NN * NN;       // (B,N)

    head_kernel<<<BB, 256, 0, stream>>>(w_key, w_beta, a_gate, w_gate,
                                        alloc, memory, prec, out_w, out_prec);

    // memory is returned unchanged: pure d2d copy overlapping on the stream
    hipMemcpyAsync(out_mem, memory, (size_t)BB * NN * WW * sizeof(float),
                   hipMemcpyDeviceToDevice, stream);

    // link update reads out_w produced by head_kernel (stream-ordered)
    link_kernel<<<BB * NN, 256, 0, stream>>>(linkm, out_w, prec, out_link);
}
